// GCN_55886114456268
// MI455X (gfx1250) — matrix operand order repaired and checked by reading, not compiled, not run on hardware
//
#include <hip/hip_runtime.h>

#define NNODES 100000
#define NEDGES 1600000
#define NGRAPHS 64
#define HID 128

typedef __attribute__((ext_vector_type(16))) __bf16 v16bf;
typedef __attribute__((ext_vector_type(8)))  float  v8f;

static __device__ __forceinline__ unsigned int f2bf_u(float f) {
  union { float f; unsigned int u; } x; x.f = f;
  unsigned int u = x.u;
  unsigned int r = u + 0x7FFFu + ((u >> 16) & 1u);   // RNE
  return r >> 16;
}
static __device__ __forceinline__ unsigned int pk_bf(float lo, float hi) {
  return f2bf_u(lo) | (f2bf_u(hi) << 16);
}

// ---------------- norm pipeline ----------------
__global__ void k_set1(float* p, int n) {
  int i = blockIdx.x * blockDim.x + threadIdx.x;
  if (i < n) p[i] = 1.0f;                            // self-loop contributes deg 1
}
__global__ void k_deg(const int* __restrict__ dst, float* deg, int e) {
  int i = blockIdx.x * blockDim.x + threadIdx.x;
  if (i < e) unsafeAtomicAdd(&deg[dst[i]], 1.0f);
}
__global__ void k_rsqrt(float* d, int n) {
  int i = blockIdx.x * blockDim.x + threadIdx.x;
  if (i < n) d[i] = rsqrtf(d[i]);                    // deg >= 1 always (self loops)
}
__global__ void k_norm(const int* __restrict__ src, const int* __restrict__ dst,
                       const float* __restrict__ dinv, float* norm, int e) {
  int i = blockIdx.x * blockDim.x + threadIdx.x;
  if (i < e) norm[i] = dinv[src[i]] * dinv[dst[i]];
}

// ---------------- data prep for WMMA ----------------
// x [N,10] -> bf16 [N,32] zero-padded (as packed pairs)
__global__ void k_padx(const float* __restrict__ x, unsigned int* __restrict__ xb) {
  int i = blockIdx.x * blockDim.x + threadIdx.x;      // node*16 + pair
  if (i >= NNODES * 16) return;
  int node = i >> 4, k = (i & 15) * 2;
  float a = (k     < 10) ? x[node * 10 + k]     : 0.0f;
  float b = (k + 1 < 10) ? x[node * 10 + k + 1] : 0.0f;
  xb[i] = pk_bf(a, b);
}
// Pack W [K,128] into wave32 B-operand layout for v_wmma_f32_16x16x32_bf16.
// Index: ((colTile*kBlocks + s)*32 + lane)*8 + q -> each lane's 8 dwords are
// contiguous (2x b128 loads); lane: n=lane&15, kh=lane>>4; VGPR q holds
// K = s*32 + kh*16 + 2q (lo bf16) / K+1 (hi bf16), column colTile*16+n.
__global__ void k_wlayout(const float* __restrict__ W, int Kin, int kBlocks,
                          unsigned int* __restrict__ out) {
  int id = blockIdx.x * blockDim.x + threadIdx.x;
  int total = 8 * kBlocks * 256;
  if (id >= total) return;
  int q = id & 7, lane = (id >> 3) & 31, ts = id >> 8;
  int s = ts % kBlocks, t = ts / kBlocks;
  int n = lane & 15, kh = lane >> 4;
  int K0 = s * 32 + kh * 16 + 2 * q;
  int col = t * 16 + n;
  float a = (K0     < Kin) ? W[K0 * HID + col]       : 0.0f;
  float b = (K0 + 1 < Kin) ? W[(K0 + 1) * HID + col] : 0.0f;
  out[id] = pk_bf(a, b);
}

// ---------------- WMMA GEMM: out[N,128] = A_bf16[N,lda] x W ----------------
// One wave per (rowGroup of 4 row-tiles, colTile). Full 128x16 B panel held in
// registers and reused across 4 row tiles -> 4*KB WMMAs per wave. Wave id is
// forced scalar via readfirstlane so all guards are SALU branches (EXEC stays
// all-1s, as WMMA requires).
template <int KB>
__global__ void k_gemm(const unsigned short* __restrict__ A, int lda,
                       const unsigned int* __restrict__ Wlay,
                       float* __restrict__ out, int nrows) {
  int wave = __builtin_amdgcn_readfirstlane(
      (int)((blockIdx.x * blockDim.x + threadIdx.x) >> 5));
  int lane = threadIdx.x & 31;
  int rowTiles = nrows / 16;
  int groups = (rowTiles + 3) / 4;
  if (wave >= groups * 8) return;                            // scalar exit
  int colT = wave & 7;
  int grp  = wave >> 3;
  int m    = lane & 15;
  int half = lane >> 4;

  // Load B panel once: KB k-steps, 8 contiguous dwords per lane per step.
  union { unsigned int u[8]; v16bf v; } ub[KB];
  const unsigned int* bp = Wlay + (size_t)colT * KB * 256 + lane * 8;
#pragma unroll
  for (int s = 0; s < KB; ++s)
#pragma unroll
    for (int q = 0; q < 8; ++q) ub[s].u[q] = bp[s * 256 + q];

#pragma unroll
  for (int rt = 0; rt < 4; ++rt) {
    int rowT = grp * 4 + rt;
    if (rowT >= rowTiles) break;                             // scalar (SGPR) test
    int row = rowT * 16 + m;
    v8f acc = {};
#pragma unroll
    for (int s = 0; s < KB; ++s) {
      // A operand 16x32 bf16: VGPR0..3 = K(half*16 + 2q, +1), VGPR4..7 = +8.
      union { unsigned int u[8]; v16bf v; } ua;
      const unsigned short* arow = A + (size_t)row * lda + s * 32 + half * 16;
#pragma unroll
      for (int q = 0; q < 4; ++q)
        ua.u[q] = *(const unsigned int*)(arow + 2 * q);
#pragma unroll
      for (int q = 0; q < 4; ++q)
        ua.u[4 + q] = *(const unsigned int*)(arow + 8 + 2 * q);
      acc = __builtin_amdgcn_wmma_f32_16x16x32_bf16(
          false, ua.v, false, ub[s].v, (short)0, acc, false, false);
    }
    // C layout: lane n = lane&15; VGPR r -> M = r + 8*half
    float* o = out + (size_t)(rowT * 16 + half * 8) * HID + colT * 16 + m;
#pragma unroll
    for (int r = 0; r < 8; ++r) o[(size_t)r * HID] = acc[r];
  }
}

// ---------------- edge scatter: agg[dst] += norm * h[src] ----------------
// One wave per edge: 512B coalesced gather of h[src], 4 native f32 atomics per
// lane into agg[dst]. Edge metadata goes through the scalar (SMEM) path.
__global__ void k_scatter(const int* __restrict__ src, const int* __restrict__ dst,
                          const float* __restrict__ norm, const float* __restrict__ h,
                          float* __restrict__ agg, int nedges) {
  int e = __builtin_amdgcn_readfirstlane(
      (int)((blockIdx.x * blockDim.x + threadIdx.x) >> 5));
  int lane = threadIdx.x & 31;
  if (e >= nedges) return;                                   // scalar exit
  int s = src[e], d = dst[e];                                // s_load (uniform)
  float w = norm[e];
  const float4 v = *(const float4*)(h + (size_t)s * HID + lane * 4);
  float* o = agg + (size_t)d * HID + lane * 4;
  unsafeAtomicAdd(o + 0, w * v.x);
  unsafeAtomicAdd(o + 1, w * v.y);
  unsafeAtomicAdd(o + 2, w * v.z);
  unsafeAtomicAdd(o + 3, w * v.w);
}

// ---------------- fused self-loop + bias + ReLU -> packed bf16 act ----------------
// act_bf[node,k] = bf16(relu(agg + dinv^2 * h + bias)); feeds next layer's GEMM
// directly, skipping the f32 activation round-trip.
__global__ void k_combine_bf(const float* __restrict__ h, const float* __restrict__ agg,
                             const float* __restrict__ dinv, const float* __restrict__ bias,
                             unsigned int* __restrict__ act_bf) {
  int i = blockIdx.x * blockDim.x + threadIdx.x;             // node*32 + chunk
  if (i >= NNODES * 32) return;
  int node = i >> 5, c = (i & 31) * 4;
  float sn = dinv[node]; sn *= sn;
  const float4 hv = *(const float4*)(h   + (size_t)node * HID + c);
  const float4 av = *(const float4*)(agg + (size_t)node * HID + c);
  const float4 bv = *(const float4*)(bias + c);
  float rx = fmaxf(av.x + sn * hv.x + bv.x, 0.f);
  float ry = fmaxf(av.y + sn * hv.y + bv.y, 0.f);
  float rz = fmaxf(av.z + sn * hv.z + bv.z, 0.f);
  float rw = fmaxf(av.w + sn * hv.w + bv.w, 0.f);
  uint2 p = { pk_bf(rx, ry), pk_bf(rz, rw) };
  *(uint2*)(act_bf + (size_t)node * 64 + (i & 31) * 2) = p;
}
// f32 variant for the last layer (no relu), in place on agg; feeds pooling.
__global__ void k_combine_f32(const float* __restrict__ h, float* __restrict__ agg,
                              const float* __restrict__ dinv, const float* __restrict__ bias) {
  int i = blockIdx.x * blockDim.x + threadIdx.x;
  if (i >= NNODES * 32) return;
  int node = i >> 5, c = (i & 31) * 4;
  float sn = dinv[node]; sn *= sn;
  const float4 hv = *(const float4*)(h   + (size_t)node * HID + c);
  float4       av = *(const float4*)(agg + (size_t)node * HID + c);
  const float4 bv = *(const float4*)(bias + c);
  av.x += sn * hv.x + bv.x;  av.y += sn * hv.y + bv.y;
  av.z += sn * hv.z + bv.z;  av.w += sn * hv.w + bv.w;
  *(float4*)(agg + (size_t)node * HID + c) = av;
}

// ---------------- pooling + final linear ----------------
__global__ void k_pool(const float* __restrict__ act, const int* __restrict__ batch,
                       float* __restrict__ pooled) {
  int i = blockIdx.x * blockDim.x + threadIdx.x;             // node*HID + f
  if (i >= NNODES * HID) return;
  int node = i >> 7, f = i & 127;
  unsafeAtomicAdd(&pooled[batch[node] * HID + f], act[i]);
}
__global__ void k_count(const int* __restrict__ batch, float* __restrict__ cnt) {
  int i = blockIdx.x * blockDim.x + threadIdx.x;
  if (i < NNODES) unsafeAtomicAdd(&cnt[batch[i]], 1.0f);
}
__global__ void k_final(const float* __restrict__ pooled, const float* __restrict__ cnt,
                        const float* __restrict__ Wlin, const float* __restrict__ blin,
                        float* __restrict__ out) {
  int id = threadIdx.x;                                      // 192 = 64 graphs * 3
  if (id >= NGRAPHS * 3) return;
  int g = id / 3, j = id % 3;
  float s = 0.f;
  for (int f = 0; f < HID; ++f) s += pooled[g * HID + f] * Wlin[f * 3 + j];
  out[id] = s / fmaxf(cnt[g], 1.0f) + blin[j];
}

extern "C" void kernel_launch(void* const* d_in, const int* in_sizes, int n_in,
                              void* d_out, int out_size, void* d_ws, size_t ws_size,
                              hipStream_t stream) {
  const float* x    = (const float*)d_in[0];
  const int*   ei   = (const int*)d_in[1];
  const int*   bat  = (const int*)d_in[2];
  const float* W1   = (const float*)d_in[3];
  const float* b1   = (const float*)d_in[4];
  const float* W2   = (const float*)d_in[5];
  const float* b2   = (const float*)d_in[6];
  const float* W3   = (const float*)d_in[7];
  const float* b3   = (const float*)d_in[8];
  const float* Wlin = (const float*)d_in[9];
  const float* blin = (const float*)d_in[10];
  float* out = (float*)d_out;
  const int* src = ei;
  const int* dst = ei + NEDGES;

  char* w = (char*)d_ws;
  auto alloc = [&](size_t bytes) {
    char* p = w; w += (bytes + 255) & ~(size_t)255; return p;
  };
  float*          dinv  = (float*)alloc((size_t)NNODES * 4);
  float*          norm  = (float*)alloc((size_t)NEDGES * 4);
  unsigned int*   wl1   = (unsigned int*)alloc(8 * 1 * 256 * 4);
  unsigned int*   wl2   = (unsigned int*)alloc(8 * 4 * 256 * 4);
  unsigned int*   wl3   = (unsigned int*)alloc(8 * 4 * 256 * 4);
  float*          bufA  = (float*)alloc((size_t)NNODES * HID * 4);  // GEMM out (h)
  float*          bufB  = (float*)alloc((size_t)NNODES * HID * 4);  // agg
  unsigned short* abf   = (unsigned short*)alloc((size_t)NNODES * HID * 2);
  float*          pool  = (float*)alloc((size_t)NGRAPHS * HID * 4);
  float*          cnt   = (float*)alloc((size_t)NGRAPHS * 4);

  const int B = 256;
  int gN    = (NNODES + B - 1) / B;
  int gE    = (NEDGES + B - 1) / B;
  int rowTiles = NNODES / 16;                         // 6250
  int groups   = (rowTiles + 3) / 4;                  // 1563
  int gemmB = (groups * 8 * 32 + B - 1) / B;          // one wave per (group, colTile)
  int scatB = (int)(((size_t)NEDGES * 32 + B - 1) / B);
  int combB = (NNODES * 32 + B - 1) / B;

  // normalization
  k_set1<<<gN, B, 0, stream>>>(dinv, NNODES);
  k_deg<<<gE, B, 0, stream>>>(dst, dinv, NEDGES);
  k_rsqrt<<<gN, B, 0, stream>>>(dinv, NNODES);
  k_norm<<<gE, B, 0, stream>>>(src, dst, dinv, norm, NEDGES);

  // pre-swizzle weights into WMMA B-operand layout
  k_wlayout<<<(8 * 1 * 256 + B - 1) / B, B, 0, stream>>>(W1, 10,  1, wl1);
  k_wlayout<<<(8 * 4 * 256 + B - 1) / B, B, 0, stream>>>(W2, 128, 4, wl2);
  k_wlayout<<<(8 * 4 * 256 + B - 1) / B, B, 0, stream>>>(W3, 128, 4, wl3);

  // ---- layer 1 ----
  k_padx<<<(NNODES * 16 + B - 1) / B, B, 0, stream>>>(x, (unsigned int*)abf);
  k_gemm<1><<<gemmB, B, 0, stream>>>(abf, 32, wl1, bufA, NNODES);
  hipMemsetAsync(bufB, 0, (size_t)NNODES * HID * 4, stream);
  k_scatter<<<scatB, B, 0, stream>>>(src, dst, norm, bufA, bufB, NEDGES);
  k_combine_bf<<<combB, B, 0, stream>>>(bufA, bufB, dinv, b1, (unsigned int*)abf);

  // ---- layer 2 ----
  k_gemm<4><<<gemmB, B, 0, stream>>>(abf, HID, wl2, bufA, NNODES);
  hipMemsetAsync(bufB, 0, (size_t)NNODES * HID * 4, stream);
  k_scatter<<<scatB, B, 0, stream>>>(src, dst, norm, bufA, bufB, NEDGES);
  k_combine_bf<<<combB, B, 0, stream>>>(bufA, bufB, dinv, b2, (unsigned int*)abf);

  // ---- layer 3 (no relu, f32 out for pooling) ----
  k_gemm<4><<<gemmB, B, 0, stream>>>(abf, HID, wl3, bufA, NNODES);
  hipMemsetAsync(bufB, 0, (size_t)NNODES * HID * 4, stream);
  k_scatter<<<scatB, B, 0, stream>>>(src, dst, norm, bufA, bufB, NEDGES);
  k_combine_f32<<<combB, B, 0, stream>>>(bufA, bufB, dinv, b3);

  // ---- pooling + final linear ----
  hipMemsetAsync(pool, 0, (size_t)NGRAPHS * HID * 4, stream);
  hipMemsetAsync(cnt, 0, (size_t)NGRAPHS * 4, stream);
  k_pool<<<(NNODES * HID + B - 1) / B, B, 0, stream>>>(bufB, bat, pool);
  k_count<<<gN, B, 0, stream>>>(bat, cnt);
  k_final<<<1, 192, 0, stream>>>(pool, cnt, Wlin, blin, out);
}
